// FineToCoarseEncoder_86225763435147
// MI455X (gfx1250) — matrix operand order repaired and checked by reading, not compiled, not run on hardware
//
#include <hip/hip_runtime.h>
#include <hip/hip_bf16.h>

// ---------------------------------------------------------------------------
// Fused 3-layer MLP (GEMM+bias+leaky x2, GEMM+bias) via v_wmma_f32_16x16x32_bf16
// 128 rows x 256 cols per 256-thread block; each wave32 owns one 16-row strip
// across all 256 columns (16 accumulator tiles, f32 regs). Double-buffered LDS
// staging: global loads for chunk k+1 are issued before the 16-WMMA burst of
// chunk k and committed to LDS after it, so L2 latency hides under the XDL ops.
//   MODE 0: edge MLP   (gather x[src]/h3[dst]/edge_attr; epilogue: atomic scatter-add)
//   MODE 1: node MLP   (gather h3/agg;                   epilogue: store nodes)
//   MODE 2: resid MLP  (gather latent_edge_attr;         epilogue: store + skip)
// ---------------------------------------------------------------------------

typedef __attribute__((ext_vector_type(16))) __bf16 v16bf;
typedef __attribute__((ext_vector_type(8)))  float  v8f;
typedef unsigned short u16;
typedef unsigned int   u32;

#define E_FINE   64800
#define NC       842
#define NBATCH   8
#define NLAT     5894
#define MROWS    128         // rows per block
#define NTILES   16          // 16x16 col tiles per wave (full 256 cols)
#define SA_STRIDE  40        // A-chunk  [128][40] bf16 (80B rows, 16B aligned)
#define SBT_STRIDE 40        // W-chunk transposed [256][40] bf16
#define SH_STRIDE  264       // H buffer [128][264] bf16 (528B rows, 16B aligned)

static __device__ __forceinline__ u16 f2bf(float f) {
  u32 u = __float_as_uint(f);
  u32 r = u + 0x7FFFu + ((u >> 16) & 1u);   // round-to-nearest-even
  return (u16)(r >> 16);
}

// A fragment: 16x32 bf16, contiguous K order (row m = lane%16, lane-half h holds K 16h..16h+15)
static __device__ __forceinline__ v16bf load_frag_a(const u16* buf, int stride,
                                                    int m0, int kb, int lane) {
  int m  = m0 + (lane & 15);
  int kg = (lane >> 4) << 4;                // 0 or 16
  const u16* p = buf + m * stride + kb + kg;
  union { u32 u[8]; v16bf v; } r;
#pragma unroll
  for (int j = 0; j < 8; ++j) {
    int off = 2 * j;
    r.u[j] = *(const u32*)(p + off);
  }
  return r.v;
}

// B fragment from transposed weight chunk [n][k]: lane n fixed, K contiguous (32B run)
static __device__ __forceinline__ v16bf load_frag_b(const u16* buf, int n0, int lane) {
  int n  = n0 + (lane & 15);
  int kh = (lane >> 4) << 4;                // 0 or 16
  const u16* p = buf + n * SBT_STRIDE + kh;
  union { u32 u[8]; v16bf v; } r;
#pragma unroll
  for (int j = 0; j < 8; ++j) r.u[j] = *(const u32*)(p + 2 * j);
  return r.v;
}

static __device__ __forceinline__ void init_bias(v8f acc[NTILES], const float* bvec,
                                                 int lane) {
  int nn = lane & 15;
#pragma unroll
  for (int t = 0; t < NTILES; ++t) {
    float bv = bvec[t * 16 + nn];
#pragma unroll
    for (int i = 0; i < 8; ++i) acc[t][i] = bv;
  }
}

static __device__ __forceinline__ void store_h(u16* sH, v8f acc[NTILES], int m0, int lane) {
  int mh = (lane >> 4) * 8;
  int nn = lane & 15;
#pragma unroll
  for (int t = 0; t < NTILES; ++t) {
    int n = t * 16 + nn;
#pragma unroll
    for (int i = 0; i < 8; ++i) {
      float v = acc[t][i];
      v = fmaxf(v, 0.01f * v);              // leaky relu, branchless
      sH[(m0 + mh + i) * SH_STRIDE + n] = f2bf(v);
    }
  }
}

// Weight chunk staging, split into (global loads -> regs) and (regs -> LDS).
static __device__ __forceinline__ void wload(uint4 w[4], const u16* __restrict__ Wt,
                                             int Kpad, int kb, int tid) {
  const u16* src = Wt + (size_t)tid * Kpad + kb;        // 16B aligned
  w[0] = *(const uint4*)(src + 0);
  w[1] = *(const uint4*)(src + 8);
  w[2] = *(const uint4*)(src + 16);
  w[3] = *(const uint4*)(src + 24);
}
static __device__ __forceinline__ void wstore(u16* sBT, const uint4 w[4], int tid) {
  uint4* dst = (uint4*)&sBT[tid * SBT_STRIDE];          // 80B rows, 16B aligned
  dst[0] = w[0]; dst[1] = w[1]; dst[2] = w[2]; dst[3] = w[3];
}

// 16 tiles of one chunk: per-tile B frag load + WMMA (XDL overlaps next DS loads)
static __device__ __forceinline__ void wmma_chunk(v8f acc[NTILES], v16bf af,
                                                  const u16* sBT, int lane) {
#pragma unroll
  for (int t = 0; t < NTILES; ++t) {
    v16bf bfg = load_frag_b(sBT, t * 16, lane);
    acc[t] = __builtin_amdgcn_wmma_f32_16x16x32_bf16(
        false, af, false, bfg, (short)0, acc[t], false, false);
  }
}

// One K=256 GEMM layer from sH with double-buffered weight staging
static __device__ __forceinline__ void run_layer256(v8f acc[NTILES], const u16* sH,
                                                    u16* sBT0, u16* sBT1,
                                                    const u16* __restrict__ Wt,
                                                    const float* __restrict__ bvec,
                                                    int tid, int lane, int m0) {
  init_bias(acc, bvec, lane);
  uint4 w[4];
  wload(w, Wt, 256, 0, tid);
  wstore(sBT0, w, tid);                     // chunk 0 staged (exposed once per layer)
#pragma unroll 1
  for (int kc = 0; kc < 8; ++kc) {
    u16* cur = (kc & 1) ? sBT1 : sBT0;
    u16* nxt = (kc & 1) ? sBT0 : sBT1;
    __syncthreads();                        // cur visible; nxt free
    bool have = (kc + 1) < 8;
    if (have) wload(w, Wt, 256, (kc + 1) * 32, tid);   // loads fly under WMMAs
    v16bf af = load_frag_a(sH, SH_STRIDE, m0, kc * 32, lane);
    wmma_chunk(acc, af, cur, lane);
    if (have) wstore(nxt, w, tid);          // commit after compute burst
  }
}

template <int MODE, int KCH0>
__global__ __launch_bounds__(256) void mlp3_fused(
    int nrows,
    const u16* __restrict__ W0t, const float* __restrict__ b0,
    const u16* __restrict__ W1t, const float* __restrict__ b1,
    const u16* __restrict__ W2t, const float* __restrict__ b2,
    const float* __restrict__ x, const int* __restrict__ edge_index,
    const float* __restrict__ edge_attr, const float* __restrict__ h3,
    const float* __restrict__ aux,          // MODE1: agg  | MODE2: latent_edge_attr
    const float* __restrict__ Ws, const float* __restrict__ bs,
    float* __restrict__ out) {
  __shared__ __align__(16) u16 sA0[MROWS * SA_STRIDE];   // 10240 B
  __shared__ __align__(16) u16 sA1[MROWS * SA_STRIDE];   // 10240 B
  __shared__ __align__(16) u16 sB0[256 * SBT_STRIDE];    // 20480 B
  __shared__ __align__(16) u16 sB1[256 * SBT_STRIDE];    // 20480 B
  __shared__ __align__(16) u16 sH[MROWS * SH_STRIDE];    // 67584 B

  const int tid   = threadIdx.x;
  const int lane  = tid & 31;
  const int wave  = tid >> 5;
  const int m0    = wave * 16;              // 8 row strips
  const int rowbase = blockIdx.x * MROWS;
  const int batch   = blockIdx.y;

  // ---- per-row gather state, hoisted out of the chunk loop ----
  const int r_ld = tid >> 1;                // gather row owner (0..127)
  const int kq   = (tid & 1) * 16;          // 16 consecutive K per thread
  const int Rg   = rowbase + r_ld;
  const bool rv  = Rg < nrows;
  const float* xrow  = nullptr;             // MODE0: x[src] row
  const float* h3row = nullptr;             // MODE0: h3[dst] row | MODE1: h3[c] row
  const float* auxrow = nullptr;            // MODE1: agg row
  float ea0 = 0.f, ea1 = 0.f;               // MODE0 edge_attr | MODE2 latent attrs
  if (rv) {
    if (MODE == 0) {
      int srcE = edge_index[Rg];
      int dstE = edge_index[E_FINE + Rg];
      xrow  = x + ((size_t)batch * E_FINE + srcE) * 78;
      h3row = h3 + (size_t)dstE * 78;
      ea0 = edge_attr[(size_t)Rg * 2];
      ea1 = edge_attr[(size_t)Rg * 2 + 1];
    } else if (MODE == 1) {
      int bb = Rg / NC, c = Rg - bb * NC;
      h3row  = h3 + (size_t)c * 78;
      auxrow = aux + ((size_t)bb * NC + c) * 256;
    } else {
      ea0 = aux[(size_t)Rg * 2];
      ea1 = aux[(size_t)Rg * 2 + 1];
    }
  }

  // gather 16 consecutive K values of this thread's row into regs
  auto gather_chunk = [&](float g[16], int kb) {
#pragma unroll
    for (int j = 0; j < 16; ++j) {
      int k = kb + kq + j;
      float v = 0.f;
      if (rv) {
        if (MODE == 0) {
          if (k < 78)        v = __builtin_nontemporal_load(&xrow[k]);
          else if (k < 156)  v = h3row[k - 78];
          else if (k == 156) v = ea0;
          else if (k == 157) v = ea1;
        } else if (MODE == 1) {
          if (k < 78)        v = h3row[k];
          else if (k < 334)  v = auxrow[k - 78];
        } else {
          if (k == 0)      v = ea0;
          else if (k == 1) v = ea1;
        }
      }
      g[j] = v;
    }
  };
  auto gstore = [&](u16* sA, const float g[16]) {
    u32 pk[8];
#pragma unroll
    for (int jj = 0; jj < 8; ++jj)
      pk[jj] = (u32)f2bf(g[2 * jj]) | ((u32)f2bf(g[2 * jj + 1]) << 16);
    uint4* dstp = (uint4*)&sA[r_ld * SA_STRIDE + kq];
    dstp[0] = *(const uint4*)&pk[0];
    dstp[1] = *(const uint4*)&pk[4];
  };

  v8f acc[NTILES];
  init_bias(acc, b0, lane);

  // ---------------- layer 0: K padded to KCH0*32, double buffered ----------------
  {
    float g[16];
    uint4 w[4];
    gather_chunk(g, 0);
    gstore(sA0, g);
    wload(w, W0t, KCH0 * 32, 0, tid);
    wstore(sB0, w, tid);
#pragma unroll 1
    for (int kc = 0; kc < KCH0; ++kc) {
      u16* curA = (kc & 1) ? sA1 : sA0;
      u16* nxtA = (kc & 1) ? sA0 : sA1;
      u16* curB = (kc & 1) ? sB1 : sB0;
      u16* nxtB = (kc & 1) ? sB0 : sB1;
      __syncthreads();
      bool have = (kc + 1) < KCH0;
      if (have) {
        gather_chunk(g, (kc + 1) * 32);     // global loads fly under WMMAs
        wload(w, W0t, KCH0 * 32, (kc + 1) * 32, tid);
      }
      v16bf af = load_frag_a(curA, SA_STRIDE, m0, 0, lane);
      wmma_chunk(acc, af, curB, lane);
      if (have) {
        gstore(nxtA, g);                    // commit after compute burst
        wstore(nxtB, w, tid);
      }
    }
  }
  __syncthreads();
  store_h(sH, acc, m0, lane);               // leaky(h0) -> sH (bf16)
  __syncthreads();

  // ---------------- layer 1 (K=256) ----------------
  run_layer256(acc, sH, sB0, sB1, W1t, b1, tid, lane, m0);
  __syncthreads();
  store_h(sH, acc, m0, lane);               // leaky(h1) -> sH
  __syncthreads();

  // ---------------- layer 2 (K=256, linear) ----------------
  run_layer256(acc, sH, sB0, sB1, W2t, b2, tid, lane, m0);

  // ---------------- epilogue (per-row hoisted) ----------------
  const int mh = (lane >> 4) * 8;
  const int nn = lane & 15;
#pragma unroll
  for (int i = 0; i < 8; ++i) {
    int R = rowbase + m0 + mh + i;
    if (R >= nrows) continue;
    if (MODE == 0) {        // scatter-add messages into agg[batch, dst, :]
      int dstE = edge_index[E_FINE + R];
      float* base = &out[((size_t)batch * NC + dstE) * 256 + nn];
#pragma unroll
      for (int t = 0; t < NTILES; ++t) unsafeAtomicAdd(&base[t * 16], acc[t][i]);
    } else if (MODE == 1) { // nodes output
      float* base = &out[(size_t)R * 256 + nn];
#pragma unroll
      for (int t = 0; t < NTILES; ++t) base[t * 16] = acc[t][i];
    } else {                // attrs output + parametrized skip
      float a0 = aux[(size_t)R * 2], a1 = aux[(size_t)R * 2 + 1];
      float* base = &out[(size_t)R * 256 + nn];
#pragma unroll
      for (int t = 0; t < NTILES; ++t) {
        int n = t * 16 + nn;
        base[t * 16] = acc[t][i] + a0 * Ws[n] + a1 * Ws[256 + n] + bs[n];
      }
    }
  }
}

// Convert one weight matrix W[K][256] (fp32) -> Wt[256][Kpad] (bf16, zero padded)
__global__ void cvt_w_bf16_t(const float* __restrict__ W, u16* __restrict__ Wt,
                             int K, int Kpad) {
  int i = blockIdx.x * blockDim.x + threadIdx.x;
  int total = 256 * Kpad;
  if (i < total) {
    int c = i / Kpad, k = i - c * Kpad;
    float v = (k < K) ? W[(size_t)k * 256 + c] : 0.f;
    Wt[i] = f2bf(v);
  }
}

__global__ void zero_f32(float* __restrict__ p, int n) {
  int i = blockIdx.x * blockDim.x + threadIdx.x;
  if (i < n) p[i] = 0.f;
}

__global__ void copy_i32(const int* __restrict__ s, int* __restrict__ d, int n) {
  int i = blockIdx.x * blockDim.x + threadIdx.x;
  if (i < n) d[i] = s[i];
}

extern "C" void kernel_launch(void* const* d_in, const int* in_sizes, int n_in,
                              void* d_out, int out_size, void* d_ws, size_t ws_size,
                              hipStream_t stream) {
  (void)in_sizes; (void)n_in; (void)out_size; (void)ws_size;
  const float* x         = (const float*)d_in[0];
  const int*   edge_idx  = (const int*)d_in[1];
  const float* edge_attr = (const float*)d_in[2];
  const int*   lat_idx   = (const int*)d_in[3];
  const float* lat_attr  = (const float*)d_in[4];
  const float* h3        = (const float*)d_in[5];
  const float* eW0 = (const float*)d_in[6];  const float* eb0 = (const float*)d_in[7];
  const float* eW1 = (const float*)d_in[8];  const float* eb1 = (const float*)d_in[9];
  const float* eW2 = (const float*)d_in[10]; const float* eb2 = (const float*)d_in[11];
  const float* nW0 = (const float*)d_in[12]; const float* nb0 = (const float*)d_in[13];
  const float* nW1 = (const float*)d_in[14]; const float* nb1 = (const float*)d_in[15];
  const float* nW2 = (const float*)d_in[16]; const float* nb2 = (const float*)d_in[17];
  const float* rW0 = (const float*)d_in[18]; const float* rb0 = (const float*)d_in[19];
  const float* rW1 = (const float*)d_in[20]; const float* rb1 = (const float*)d_in[21];
  const float* rW2 = (const float*)d_in[22]; const float* rb2 = (const float*)d_in[23];
  const float* rWs = (const float*)d_in[24]; const float* rbs = (const float*)d_in[25];

  float* out       = (float*)d_out;
  float* nodes_out = out;                                  // [8*842, 256]
  int*   idx_out   = (int*)(out + (size_t)NBATCH * NC * 256);
  float* attrs_out = out + (size_t)NBATCH * NC * 256 + 2 * NLAT;

  // ---- workspace layout ----
  float* agg = (float*)d_ws;                               // [8, 842, 256] f32
  u16*   wbase = (u16*)((char*)d_ws + (size_t)NBATCH * NC * 256 * sizeof(float));
  // bf16 transposed weights, zero padded K: [256][Kpad]
  u16* eW0t = wbase;                 // 256*160
  u16* eW1t = eW0t + 256 * 160;      // 256*256
  u16* eW2t = eW1t + 256 * 256;
  u16* nW0t = eW2t + 256 * 256;      // 256*352
  u16* nW1t = nW0t + 256 * 352;
  u16* nW2t = nW1t + 256 * 256;
  u16* rW0t = nW2t + 256 * 256;      // 256*32
  u16* rW1t = rW0t + 256 * 32;
  u16* rW2t = rW1t + 256 * 256;

  // ---- weight pre-conversion (bf16, transposed, padded) ----
  struct { const float* W; u16* Wt; int K, Kpad; } cv[9] = {
      {eW0, eW0t, 158, 160}, {eW1, eW1t, 256, 256}, {eW2, eW2t, 256, 256},
      {nW0, nW0t, 334, 352}, {nW1, nW1t, 256, 256}, {nW2, nW2t, 256, 256},
      {rW0, rW0t, 2, 32},    {rW1, rW1t, 256, 256}, {rW2, rW2t, 256, 256}};
  for (int i = 0; i < 9; ++i) {
    int total = 256 * cv[i].Kpad;
    cvt_w_bf16_t<<<dim3((total + 255) / 256), dim3(256), 0, stream>>>(
        cv[i].W, cv[i].Wt, cv[i].K, cv[i].Kpad);
  }

  const int nagg = NBATCH * NC * 256;
  zero_f32<<<dim3((nagg + 255) / 256), dim3(256), 0, stream>>>(agg, nagg);

  // edge MLP: K0 = 158 -> 5 chunks; rows per batch = 64800; scatter into agg
  mlp3_fused<0, 5><<<dim3((E_FINE + MROWS - 1) / MROWS, NBATCH), dim3(256), 0, stream>>>(
      E_FINE, eW0t, eb0, eW1t, eb1, eW2t, eb2,
      x, edge_idx, edge_attr, h3, nullptr, nullptr, nullptr, agg);

  // node MLP: K0 = 334 -> 11 chunks; rows = 8*842
  const int nrows1 = NBATCH * NC;
  mlp3_fused<1, 11><<<dim3((nrows1 + MROWS - 1) / MROWS, 1), dim3(256), 0, stream>>>(
      nrows1, nW0t, nb0, nW1t, nb1, nW2t, nb2,
      nullptr, nullptr, nullptr, h3, agg, nullptr, nullptr, nodes_out);

  // residual MLP: K0 = 2 -> 1 chunk; rows = 5894; + skip
  mlp3_fused<2, 1><<<dim3((NLAT + MROWS - 1) / MROWS, 1), dim3(256), 0, stream>>>(
      NLAT, rW0t, rb0, rW1t, rb1, rW2t, rb2,
      nullptr, nullptr, nullptr, nullptr, lat_attr, rWs, rbs, attrs_out);

  // latent_edge_index pass-through (bit-exact int copy into output region)
  copy_i32<<<dim3((2 * NLAT + 255) / 256), dim3(256), 0, stream>>>(lat_idx, idx_out, 2 * NLAT);
}
